// GlobalRetriever_5729486373216
// MI455X (gfx1250) — matrix operand order repaired and checked by reading, not compiled, not run on hardware
//
#include <hip/hip_runtime.h>
#include <math.h>

// Problem constants (from setup_inputs)
#define BQ     1024      // queries
#define NC     7         // channels
#define LT     96        // time length
#define D      672       // NC*LT flattened dim
#define NREF   50000     // reference rows
#define NPAD   50048     // NREF rounded up to multiple of 128 (zero-padded)
#define TOPK   20
#define QCHUNK 128       // queries per chunk (sims chunk = 128*50048*4 = 25.6MB, L2-resident)

// Workspace layout (total ~94.3 MB):
//   rn  : NPAD * D f16   = 67,264,512 B   (encoded+L2-normalized refs, zero-padded)
//   qn  : BQ   * D f16   =  1,376,256 B   (encoded+L2-normalized queries)
//   sims: QCHUNK*NPAD f32= 25,624,576 B   (per-chunk similarity tile)

typedef _Float16 v16h __attribute__((ext_vector_type(16)));
typedef _Float16 h8   __attribute__((ext_vector_type(8)));
typedef float    v8f  __attribute__((ext_vector_type(8)));

// ---------------------------------------------------------------------------
// Encode: per-channel z-norm over time (unbiased std + 1e-6), then L2-normalize
// the flat 672-vector (x / max(||x||, 1e-12)), emit f16. One wave per row.
// Rows in [rows, rows_pad) are zero-filled padding.
// ---------------------------------------------------------------------------
__global__ void __launch_bounds__(256) encode_kernel(const float* __restrict__ src,
                                                     _Float16* __restrict__ dst,
                                                     int rows, int rows_pad)
{
    const int row  = blockIdx.x * 8 + (threadIdx.x >> 5);
    const int lane = threadIdx.x & 31;
    if (row >= rows_pad) return;

    _Float16* out = dst + (size_t)row * D;
    if (row >= rows) {                      // zero padding rows
        for (int i = lane; i < D; i += 32) out[i] = (_Float16)0.f;
        return;
    }

    const float* a = src + (size_t)row * D;
    float v[21];                            // 3 elems per lane per channel
    float l2 = 0.f;
    #pragma unroll
    for (int c = 0; c < NC; ++c) {
        const int base = c * LT + lane * 3;
        float a0 = a[base + 0];
        float a1 = a[base + 1];
        float a2 = a[base + 2];
        float s  = a0 + a1 + a2;
        float ss = a0*a0 + a1*a1 + a2*a2;
        #pragma unroll
        for (int m = 16; m >= 1; m >>= 1) {
            s  += __shfl_xor(s,  m, 32);
            ss += __shfl_xor(ss, m, 32);
        }
        float mean = s * (1.f / 96.f);
        float var  = (ss - 96.f * mean * mean) * (1.f / 95.f);   // ddof=1
        float stdv = sqrtf(fmaxf(var, 0.f)) + 1e-6f;
        float inv  = 1.f / stdv;
        float n0 = (a0 - mean) * inv;
        float n1 = (a1 - mean) * inv;
        float n2 = (a2 - mean) * inv;
        v[c*3+0] = n0; v[c*3+1] = n1; v[c*3+2] = n2;
        l2 += n0*n0 + n1*n1 + n2*n2;
    }
    #pragma unroll
    for (int m = 16; m >= 1; m >>= 1) l2 += __shfl_xor(l2, m, 32);
    float scale = 1.f / fmaxf(sqrtf(l2), 1e-12f);
    #pragma unroll
    for (int c = 0; c < NC; ++c) {
        const int base = c * LT + lane * 3;
        out[base + 0] = (_Float16)(v[c*3+0] * scale);
        out[base + 1] = (_Float16)(v[c*3+1] * scale);
        out[base + 2] = (_Float16)(v[c*3+2] * scale);
    }
}

// ---------------------------------------------------------------------------
// Similarity GEMM: sims[m, n] = qn[m,:] . rn[n,:]  via v_wmma_f32_16x16x32_f16.
// One wave computes a 32(query) x 64(ref) tile: 2 A frags x 4 B frags -> 8 WMMAs
// per K-step on 12 b128 loads (1.5 loads/WMMA). Block = 8 waves covering a
// 128x128 tile (4 m-subtiles x 2 n-subtiles) for A/B reuse in L0.
// Fully unrolled K loop (21 steps, 168 WMMAs) keeps everything in SSA form so
// the scheduler can hoist next-step loads above current-step WMMAs
// (software pipelining; avoids s_wait_loadcnt 0 before every WMMA).
//
// ISA fragment layouts (CDNA5 ISA):
//   A 16x32 f16: lane l holds M=l&15; K-halves = k0 + 8*(l>=16) + {0..7} and +16.
//   B 32x16 f16: lane l holds N=l&15; K-halves = k0 + 8*(l>=16) + {0..7} and +16.
// Both are contiguous 16B-aligned runs of the row-major f16 matrices -> direct
// global b128 loads, no LDS transpose needed (B = rn^T for free).
// ---------------------------------------------------------------------------
__global__ void __launch_bounds__(256) sim_gemm_kernel(const _Float16* __restrict__ qn,
                                                       const _Float16* __restrict__ rn,
                                                       float* __restrict__ sims,
                                                       int qbase)
{
    const int wave = threadIdx.x >> 5;
    const int lane = threadIdx.x & 31;
    const int hi   = lane >> 4;             // 0/1 lane half-group
    const int lo   = lane & 15;
    const int mt   = wave & 3;              // 0..3 -> 32-query sub-tile
    const int ntw  = wave >> 2;             // 0..1 -> 64-ref sub-tile
    const int m0   = qbase + mt * 32;
    const int n0   = blockIdx.x * 128 + ntw * 64;

    const _Float16* aptr0 = qn + (size_t)(m0 + lo) * D + hi * 8;
    const _Float16* aptr1 = aptr0 + (size_t)16 * D;
    const _Float16* bptr  = rn + (size_t)(n0 + lo) * D + hi * 8;

    v8f acc[2][4];
    #pragma unroll
    for (int m = 0; m < 2; ++m)
        #pragma unroll
        for (int t = 0; t < 4; ++t)
            #pragma unroll
            for (int i = 0; i < 8; ++i) acc[m][t][i] = 0.f;

    #pragma unroll
    for (int ks = 0; ks < D / 32; ++ks) {   // 21 K-steps of 32, fully unrolled
        const int k0 = ks * 32;

        // ---- load all fragments for this K-step first ----
        v16h a[2], b[4];
        {
            h8 l0 = *(const h8*)(aptr0 + k0);
            h8 h0 = *(const h8*)(aptr0 + k0 + 16);
            h8 l1 = *(const h8*)(aptr1 + k0);
            h8 h1 = *(const h8*)(aptr1 + k0 + 16);
            #pragma unroll
            for (int i = 0; i < 8; ++i) {
                a[0][i] = l0[i]; a[0][i + 8] = h0[i];
                a[1][i] = l1[i]; a[1][i + 8] = h1[i];
            }
        }
        #pragma unroll
        for (int t = 0; t < 4; ++t) {
            const _Float16* bp = bptr + (size_t)t * 16 * D + k0;
            h8 bl = *(const h8*)(bp);
            h8 bh = *(const h8*)(bp + 16);
            #pragma unroll
            for (int i = 0; i < 8; ++i) { b[t][i] = bl[i]; b[t][i + 8] = bh[i]; }
        }

        // ---- 8 WMMAs on the loaded fragments ----
        #pragma unroll
        for (int m = 0; m < 2; ++m)
            #pragma unroll
            for (int t = 0; t < 4; ++t)
                acc[m][t] = __builtin_amdgcn_wmma_f32_16x16x32_f16(
                    false, a[m], false, b[t], (short)0, acc[m][t], false, false);
    }

    // C/D layout: lane l, VGPR r -> row M = r + 8*(l>=16), col N = l&15
    #pragma unroll
    for (int m = 0; m < 2; ++m) {
        const int mloc = mt * 32 + m * 16 + hi * 8;
        #pragma unroll
        for (int t = 0; t < 4; ++t)
            #pragma unroll
            for (int r = 0; r < 8; ++r)
                sims[(size_t)(mloc + r) * NPAD + (n0 + t * 16 + lo)] = acc[m][t][r];
    }
}

// ---------------------------------------------------------------------------
// Top-k + softmax + gather: one block (256 threads) per query.
// Key packing: orderable f32 in high 32 bits, (0xFFFFFFFF - idx) in low 32
// -> u64 max = largest value, ties resolved to smallest index (torch/jax order).
// ---------------------------------------------------------------------------
__global__ void __launch_bounds__(256) topk_kernel(const float* __restrict__ sims,
                                                   const float* __restrict__ ytrain,
                                                   float* __restrict__ out,
                                                   int qbase)
{
    const int tid = threadIdx.x;
    const int b   = qbase + blockIdx.x;
    const float* row = sims + (size_t)blockIdx.x * NPAD;

    // per-thread register-resident sorted top-20
    unsigned long long loc[TOPK];
    #pragma unroll
    for (int j = 0; j < TOPK; ++j) loc[j] = 0ull;

    for (int n = tid; n < NREF; n += 256) {
        float vv = row[n];
        unsigned u = __float_as_uint(vv);
        u = (u & 0x80000000u) ? ~u : (u | 0x80000000u);
        unsigned long long key = ((unsigned long long)u << 32)
                               | (unsigned long long)(0xFFFFFFFFu - (unsigned)n);
        if (key > loc[TOPK - 1]) {
            loc[TOPK - 1] = key;
            #pragma unroll
            for (int p = TOPK - 1; p > 0; --p) {
                if (loc[p] > loc[p - 1]) {
                    unsigned long long t = loc[p]; loc[p] = loc[p - 1]; loc[p - 1] = t;
                }
            }
        }
    }

    __shared__ unsigned long long cand[256 * TOPK];   // 40 KB
    #pragma unroll
    for (int j = 0; j < TOPK; ++j) cand[tid * TOPK + j] = loc[j];
    __syncthreads();

    __shared__ unsigned long long red[256];
    __shared__ float svals[TOPK];
    __shared__ int   sidx[TOPK];

    for (int r = 0; r < TOPK; ++r) {
        unsigned long long best = 0ull;
        for (int i = tid; i < 256 * TOPK; i += 256) {
            unsigned long long k = cand[i];
            if (k > best) best = k;
        }
        red[tid] = best;
        __syncthreads();
        for (int s = 128; s > 0; s >>= 1) {
            if (tid < s && red[tid + s] > red[tid]) red[tid] = red[tid + s];
            __syncthreads();
        }
        best = red[0];
        __syncthreads();
        if (tid == 0) {
            unsigned u = (unsigned)(best >> 32);
            u = (u & 0x80000000u) ? (u & 0x7FFFFFFFu) : ~u;
            svals[r] = __uint_as_float(u);
            sidx[r]  = (int)(0xFFFFFFFFu - (unsigned)(best & 0xFFFFFFFFull));
        }
        for (int i = tid; i < 256 * TOPK; i += 256)
            if (cand[i] == best) cand[i] = 0ull;      // keys unique (idx embedded)
        __syncthreads();
    }

    // softmax over the 20 (descending) values -> weights output
    if (tid == 0) {
        float mx = svals[0];
        float e[TOPK];
        float sum = 0.f;
        #pragma unroll
        for (int j = 0; j < TOPK; ++j) { e[j] = expf(svals[j] - mx); sum += e[j]; }
        float inv = 1.f / sum;
        #pragma unroll
        for (int j = 0; j < TOPK; ++j) out[(size_t)b * TOPK + j] = e[j] * inv;
    }
    __syncthreads();

    // gather Y_train[idx] -> Yk; rows are 672 floats = 168 float4 (16B aligned)
    float4* yout = (float4*)(out + (size_t)BQ * TOPK + (size_t)b * TOPK * D);
    for (int j = 0; j < TOPK; ++j) {
        const float4* srcp = (const float4*)(ytrain + (size_t)sidx[j] * D);
        for (int e2 = tid; e2 < D / 4; e2 += 256)
            yout[(size_t)j * (D / 4) + e2] = srcp[e2];
    }
}

// ---------------------------------------------------------------------------
extern "C" void kernel_launch(void* const* d_in, const int* in_sizes, int n_in,
                              void* d_out, int out_size, void* d_ws, size_t ws_size,
                              hipStream_t stream)
{
    (void)in_sizes; (void)n_in; (void)out_size; (void)ws_size;
    const float* x   = (const float*)d_in[0];   // [1024, 7, 96]
    const float* Xtr = (const float*)d_in[1];   // [50000, 7, 96]
    const float* Ytr = (const float*)d_in[2];   // [50000, 96, 7]
    // d_in[3] = topk (== 20, fixed)
    float* out = (float*)d_out;                 // [1024*20] ++ [1024*20*96*7]

    char* ws = (char*)d_ws;
    _Float16* rn   = (_Float16*)(ws);
    _Float16* qn   = (_Float16*)(ws + (size_t)NPAD * D * 2);
    float*    sims = (float*)   (ws + (size_t)NPAD * D * 2 + (size_t)BQ * D * 2);

    // Encode refs (zero-padded to NPAD rows) and queries -> f16.
    encode_kernel<<<NPAD / 8, 256, 0, stream>>>(Xtr, rn, NREF, NPAD);
    encode_kernel<<<BQ / 8,   256, 0, stream>>>(x,   qn, BQ,   BQ);

    // Chunked GEMM + top-k: sims chunk (25.6MB) stays L2-resident between the
    // producing GEMM and the consuming top-k; rn (67MB f16) is L2-resident
    // across all 8 chunks (192MB L2).
    for (int qbase = 0; qbase < BQ; qbase += QCHUNK) {
        sim_gemm_kernel<<<NPAD / 128, 256, 0, stream>>>(qn, rn, sims, qbase);
        topk_kernel<<<QCHUNK, 256, 0, stream>>>(sims, Ytr, out, qbase);
    }
}
